// TFGWPooling_13726715478794
// MI455X (gfx1250) — matrix operand order repaired and checked by reading, not compiled, not run on hardware
//
#include <hip/hip_runtime.h>

// ---- CDNA5 WMMA types ----
typedef __attribute__((ext_vector_type(16))) _Float16 v16h;
typedef __attribute__((ext_vector_type(8)))  _Float16 v8h;
typedef __attribute__((ext_vector_type(8)))  float    v8f;

#define BG 128
#define NN 256
#define DD 128
#define KT 16
#define MM 16
#define TT_LD 264          // f16 row stride for B-operand staging
#define GT_LD 17           // f32 row stride for WMMA output / negC
#define N_CG 10
#define N_SINK 50

// ---------------- prep kernels ----------------
__global__ void zero_u32_k(unsigned* __restrict__ p, int n) {
  int i = blockIdx.x * blockDim.x + threadIdx.x;
  if (i < n) p[i] = 0u;
}
__global__ void scatter_edges_k(const int* __restrict__ ei, unsigned* __restrict__ Cc, int nE) {
  int e = blockIdx.x * blockDim.x + threadIdx.x;
  if (e >= nE) return;
  int r = ei[e];           // edge_index[0][e]
  int c = ei[nE + e];      // edge_index[1][e]
  int b = r >> 8;
  atomicAdd(&Cc[((size_t)b << 16) + ((r & 255) << 8) + (c & 255)], 1u);
}
__global__ void cvt_count_f16_k(const unsigned* __restrict__ Cc, _Float16* __restrict__ Ch, int n) {
  int i = blockIdx.x * blockDim.x + threadIdx.x;
  if (i < n) Ch[i] = (_Float16)(float)Cc[i];
}
__global__ void cvt_f32_f16_k(const float* __restrict__ X, _Float16* __restrict__ Xh, int n) {
  int i = blockIdx.x * blockDim.x + threadIdx.x;
  if (i < n) Xh[i] = (_Float16)X[i];
}

// ---------------- WMMA helper ----------------
// Gtmp[256,16] (LDS, stride GT_LD) = A[256, 32*ksteps] (global f16, row-major, stride lda)
//                                    @ Tt[16, 32*ksteps]^T (LDS f16, stride TT_LD; row n = column n of B)
// 8 waves, wave w computes M-tiles 2w, 2w+1. Must be called by all 256 threads (EXEC all ones).
__device__ __forceinline__ void wmma_gemm(const _Float16* __restrict__ A, int lda, int ksteps,
                                          const _Float16* Tt, float* Gtmp, int tid)
{
  const int lane = tid & 31;
  const int wave = tid >> 5;
  const int nn   = lane & 15;
  const int hi   = lane >> 4;
  const int r0   = wave * 32 + nn;        // A-fragment row, tile 0
  const int r1   = r0 + 16;               // tile 1
  v8f acc0 = {0.f,0.f,0.f,0.f,0.f,0.f,0.f,0.f};
  v8f acc1 = {0.f,0.f,0.f,0.f,0.f,0.f,0.f,0.f};
  for (int ks = 0; ks < ksteps; ++ks) {
    const int kb = ks << 5;
    // B fragment: lane-half hi holds K = kb+8*hi..+7 and kb+8*hi+16..+23 of column nn
    const v8h* bp = (const v8h*)(Tt + nn * TT_LD + kb + (hi << 3));
    v8h b0 = bp[0], b1 = bp[2];
    v16h bv = __builtin_shufflevector(b0, b1, 0,1,2,3,4,5,6,7,8,9,10,11,12,13,14,15);
    // A fragment: lane-half hi holds K = kb+8*hi..+7 and kb+8*hi+16..+23 of its row
    const v8h* ap0 = (const v8h*)(A + (size_t)r0 * lda + kb + (hi << 3));
    v8h a00 = ap0[0], a01 = ap0[2];
    v16h av0 = __builtin_shufflevector(a00, a01, 0,1,2,3,4,5,6,7,8,9,10,11,12,13,14,15);
    const v8h* ap1 = (const v8h*)(A + (size_t)r1 * lda + kb + (hi << 3));
    v8h a10 = ap1[0], a11 = ap1[2];
    v16h av1 = __builtin_shufflevector(a10, a11, 0,1,2,3,4,5,6,7,8,9,10,11,12,13,14,15);
    acc0 = __builtin_amdgcn_wmma_f32_16x16x32_f16(false, av0, false, bv, (short)0, acc0, false, false);
    acc1 = __builtin_amdgcn_wmma_f32_16x16x32_f16(false, av1, false, bv, (short)0, acc1, false, false);
  }
  // C/D layout: lane holds column nn, rows v + 8*hi (v = VGPR index)
  #pragma unroll
  for (int v = 0; v < 8; ++v) {
    Gtmp[(wave * 32 + hi * 8 + v) * GT_LD + nn]      = acc0[v];
    Gtmp[(wave * 32 + 16 + hi * 8 + v) * GT_LD + nn] = acc1[v];
  }
}

// ---------------- main kernel: one workgroup per (b,k) ----------------
__global__ __launch_bounds__(256) void tfgw_main_k(
    const _Float16* __restrict__ Xh,     // [B*N, D] f16
    const _Float16* __restrict__ Ch,     // [B, N, N] f16 adjacency
    const float*    __restrict__ X,      // [B*N, D] f32
    const float*    __restrict__ Adj,    // [K, 16, 16]
    const float*    __restrict__ Feat,   // [K, 16, D]
    const float*    __restrict__ q0,     // [K, 16]
    const float*    __restrict__ alpha0, // [1]
    float*          __restrict__ out)    // [B, K]
{
  __shared__ float s_negC[NN * GT_LD];
  __shared__ float s_Gt[NN * GT_LD];
  __shared__ __align__(16) _Float16 s_Tt[MM * TT_LD];
  __shared__ float s_C2[MM * GT_LD];
  __shared__ float s_F[NN];
  __shared__ float s_Gv[MM];
  __shared__ float s_pm[NN];
  __shared__ float s_ps[NN];
  __shared__ float4 s_red[NN];
  __shared__ float s_c2sq[MM];
  __shared__ float s_f2[MM];
  __shared__ float s_scalar[2];
  __shared__ float4 s_sum;

  const int tid = threadIdx.x;
  const int b = blockIdx.x >> 4;
  const int k = blockIdx.x & 15;
  const float LOGP = -5.545177444479562f;  // log(1/256)

  const float al  = 1.f / (1.f + __expf(-alpha0[0]));  // sigmoid(alpha0)
  const float oma = 1.f - al;

  // q = softmax(q0[k]); per-thread copies
  float q0v[16];
  #pragma unroll
  for (int n = 0; n < 16; ++n) q0v[n] = q0[k * 16 + n];
  float qmax = q0v[0];
  #pragma unroll
  for (int n = 1; n < 16; ++n) qmax = fmaxf(qmax, q0v[n]);
  float qsum = 0.f, qv[16];
  #pragma unroll
  for (int n = 0; n < 16; ++n) { qv[n] = __expf(q0v[n] - qmax); qsum += qv[n]; }
  const float inv_qs = 1.f / qsum;
  #pragma unroll
  for (int n = 0; n < 16; ++n) qv[n] *= inv_qs;
  const float lq_mine = q0v[tid & 15] - qmax - __logf(qsum); // logq[tid&15]

  // T0 = p q^T
  float T_reg[16];
  #pragma unroll
  for (int n = 0; n < 16; ++n) T_reg[n] = qv[n] * (1.f / 256.f);

  // stage C2 (one element / thread)
  s_C2[(tid >> 4) * GT_LD + (tid & 15)] = Adj[k * 256 + tid];
  // stage f16 template features into Tt rows (B operand for X @ F^T)
  {
    int n = tid >> 4, c0 = (tid & 15) * 8;
    const float* fr = Feat + (size_t)k * MM * DD + n * DD + c0;
    #pragma unroll
    for (int u = 0; u < 8; ++u) s_Tt[n * TT_LD + c0 + u] = (_Float16)fr[u];
  }
  // x2 = |X[b,tid]|^2
  float x2 = 0.f;
  {
    const float* xr = X + ((size_t)b * NN + tid) * DD;
    for (int d = 0; d < DD; ++d) { float v = xr[d]; x2 += v * v; }
  }
  // c1sq = (C1^2 @ p)[tid]
  float c1sq = 0.f;
  {
    const v8h* cr = (const v8h*)(Ch + ((size_t)b << 16) + (size_t)tid * NN);
    for (int c = 0; c < 32; ++c) {
      v8h h = cr[c];
      #pragma unroll
      for (int e = 0; e < 8; ++e) { float v = (float)h[e]; c1sq += v * v; }
    }
    c1sq *= (1.f / 256.f);
  }
  __syncthreads();
  wmma_gemm(Xh + (size_t)b * NN * DD, DD, DD / 32, s_Tt, s_Gt, tid);
  if (tid < 16) {
    float s = 0.f;
    #pragma unroll
    for (int j = 0; j < 16; ++j) { float cv = s_C2[tid * GT_LD + j]; s += cv * cv * qv[j]; }
    s_c2sq[tid] = s;
    float f2 = 0.f;
    const float* fr = Feat + (size_t)k * MM * DD + tid * DD;
    for (int d = 0; d < DD; ++d) { float v = fr[d]; f2 += v * v; }
    s_f2[tid] = f2;
  }
  __syncthreads();

  // Mcost row (registers)
  float M_reg[16];
  #pragma unroll
  for (int n = 0; n < 16; ++n)
    M_reg[n] = x2 + s_f2[n] - 2.f * s_Gt[tid * GT_LD + n];

  const _Float16* Cb = Ch + ((size_t)b << 16);
  float W1r[16], dTr[16], ncr[16];

  for (int it = 0; it < N_CG; ++it) {
    // ---- gradient: W1 = C1 @ T @ C2^T ----
    __syncthreads();
    #pragma unroll
    for (int n = 0; n < 16; ++n) s_Tt[n * TT_LD + tid] = (_Float16)T_reg[n];
    __syncthreads();
    wmma_gemm(Cb, NN, NN / 32, s_Tt, s_Gt, tid);
    __syncthreads();
    float gr[16];
    #pragma unroll
    for (int j = 0; j < 16; ++j) gr[j] = s_Gt[tid * GT_LD + j];
    float lmax = 0.f;
    #pragma unroll
    for (int n = 0; n < 16; ++n) {
      float s = 0.f;
      #pragma unroll
      for (int j = 0; j < 16; ++j) s += gr[j] * s_C2[n * GT_LD + j];
      W1r[n] = s;
      float cost = oma * M_reg[n] + 2.f * al * (c1sq + s_c2sq[n] - 2.f * s);
      lmax = fmaxf(lmax, fabsf(cost));
    }
    s_pm[tid] = lmax;
    __syncthreads();
    if (tid == 0) {
      float m = 0.f;
      for (int i = 0; i < NN; ++i) m = fmaxf(m, s_pm[i]);
      s_scalar[0] = 0.01f * m + 1e-12f;
    }
    __syncthreads();
    {
      float inv_reg = 1.f / s_scalar[0];
      #pragma unroll
      for (int n = 0; n < 16; ++n) {
        float cost = oma * M_reg[n] + 2.f * al * (c1sq + s_c2sq[n] - 2.f * W1r[n]);
        float v = -cost * inv_reg;
        ncr[n] = v;
        s_negC[tid * GT_LD + n] = v;
      }
    }
    if (tid < 16) s_Gv[tid] = 0.f;

    // ---- log-domain Sinkhorn ----
    const int jj = tid & 15, sg = tid >> 4;
    for (int si = 0; si < N_SINK; ++si) {
      __syncthreads();                       // G (and negC on first iter) visible
      float m = -1e30f;
      #pragma unroll
      for (int n = 0; n < 16; ++n) m = fmaxf(m, ncr[n] + s_Gv[n]);
      float s = 0.f;
      #pragma unroll
      for (int n = 0; n < 16; ++n) s += __expf(ncr[n] + s_Gv[n] - m);
      s_F[tid] = LOGP - m - __logf(s);
      __syncthreads();
      float mm = -1e30f;
      #pragma unroll
      for (int r = 0; r < 16; ++r) {
        int i = sg * 16 + r;
        mm = fmaxf(mm, s_negC[i * GT_LD + jj] + s_F[i]);
      }
      s_pm[tid] = mm;
      __syncthreads();
      float cm = -1e30f;
      #pragma unroll
      for (int s2 = 0; s2 < 16; ++s2) cm = fmaxf(cm, s_pm[s2 * 16 + jj]);
      float ss = 0.f;
      #pragma unroll
      for (int r = 0; r < 16; ++r) {
        int i = sg * 16 + r;
        ss += __expf(s_negC[i * GT_LD + jj] + s_F[i] - cm);
      }
      s_ps[tid] = ss;
      __syncthreads();
      if (tid < 16) {
        float cm2 = -1e30f, st = 0.f;
        #pragma unroll
        for (int s2 = 0; s2 < 16; ++s2) {
          cm2 = fmaxf(cm2, s_pm[s2 * 16 + tid]);
          st += s_ps[s2 * 16 + tid];
        }
        s_Gv[tid] = lq_mine - cm2 - __logf(st);
      }
    }
    __syncthreads();

    // ---- dT = Tnew - T ----
    #pragma unroll
    for (int n = 0; n < 16; ++n) {
      float tn = __expf(ncr[n] + s_F[tid] + s_Gv[n]);
      dTr[n] = tn - T_reg[n];
    }
    // ---- line search: C1dC2 = C1 @ dT @ C2^T ----
    #pragma unroll
    for (int n = 0; n < 16; ++n) s_Tt[n * TT_LD + tid] = (_Float16)dTr[n];
    __syncthreads();
    wmma_gemm(Cb, NN, NN / 32, s_Tt, s_Gt, tid);
    __syncthreads();
    float aL = 0.f, b1 = 0.f, b2 = 0.f, bm = 0.f;
    #pragma unroll
    for (int j = 0; j < 16; ++j) gr[j] = s_Gt[tid * GT_LD + j];
    #pragma unroll
    for (int n = 0; n < 16; ++n) {
      float cd = 0.f;
      #pragma unroll
      for (int j = 0; j < 16; ++j) cd += gr[j] * s_C2[n * GT_LD + j];
      aL += cd * dTr[n];
      b1 += cd * T_reg[n];
      b2 += W1r[n] * dTr[n];
      bm += M_reg[n] * dTr[n];
    }
    s_red[tid] = make_float4(aL, b1, b2, bm);
    __syncthreads();
    if (tid < 64) {
      float4 v = s_red[tid], v1 = s_red[tid + 64], v2 = s_red[tid + 128], v3 = s_red[tid + 192];
      v.x += v1.x + v2.x + v3.x; v.y += v1.y + v2.y + v3.y;
      v.z += v1.z + v2.z + v3.z; v.w += v1.w + v2.w + v3.w;
      s_red[tid] = v;
    }
    __syncthreads();
    if (tid == 0) {
      float sx = 0.f, sy = 0.f, sz = 0.f, sw = 0.f;
      for (int i = 0; i < 64; ++i) {
        float4 v = s_red[i];
        sx += v.x; sy += v.y; sz += v.z; sw += v.w;
      }
      s_sum = make_float4(sx, sy, sz, sw);
    }
    __syncthreads();
    {
      float4 S = s_sum;
      float av = -2.f * al * S.x;
      float bv = oma * S.w - 2.f * al * (S.y + S.z);
      float tstep;
      if (av > 0.f) {
        tstep = -bv / (2.f * av + 1e-16f);
        tstep = fminf(1.f, fmaxf(0.f, tstep));
      } else {
        tstep = (av + bv < 0.f) ? 1.f : 0.f;
      }
      #pragma unroll
      for (int n = 0; n < 16; ++n) T_reg[n] += tstep * dTr[n];
    }
  }

  // ---- final: gw = sum(tens(T)*T), lin = sum(M*T) ----
  __syncthreads();
  #pragma unroll
  for (int n = 0; n < 16; ++n) s_Tt[n * TT_LD + tid] = (_Float16)T_reg[n];
  __syncthreads();
  wmma_gemm(Cb, NN, NN / 32, s_Tt, s_Gt, tid);
  __syncthreads();
  float gwl = 0.f, linl = 0.f;
  {
    float gr[16];
    #pragma unroll
    for (int j = 0; j < 16; ++j) gr[j] = s_Gt[tid * GT_LD + j];
    #pragma unroll
    for (int n = 0; n < 16; ++n) {
      float w = 0.f;
      #pragma unroll
      for (int j = 0; j < 16; ++j) w += gr[j] * s_C2[n * GT_LD + j];
      gwl  += (c1sq + s_c2sq[n] - 2.f * w) * T_reg[n];
      linl += M_reg[n] * T_reg[n];
    }
  }
  s_red[tid] = make_float4(gwl, linl, 0.f, 0.f);
  __syncthreads();
  if (tid < 64) {
    float4 v = s_red[tid], v1 = s_red[tid + 64], v2 = s_red[tid + 128], v3 = s_red[tid + 192];
    v.x += v1.x + v2.x + v3.x; v.y += v1.y + v2.y + v3.y;
    s_red[tid] = v;
  }
  __syncthreads();
  if (tid == 0) {
    float gw = 0.f, lin = 0.f;
    for (int i = 0; i < 64; ++i) { gw += s_red[i].x; lin += s_red[i].y; }
    out[b * KT + k] = oma * lin + al * gw;
  }
}

// ---------------- launcher ----------------
extern "C" void kernel_launch(void* const* d_in, const int* in_sizes, int n_in,
                              void* d_out, int out_size, void* d_ws, size_t ws_size,
                              hipStream_t stream) {
  const float* X      = (const float*)d_in[0];
  const int*   EI     = (const int*)  d_in[1];
  // d_in[2] (batch) unused: equal-size sorted blocks per reference
  const float* Adj    = (const float*)d_in[3];
  const float* Feat   = (const float*)d_in[4];
  const float* q0     = (const float*)d_in[5];
  const float* alpha0 = (const float*)d_in[6];
  float* out = (float*)d_out;

  // workspace layout: Cc(u32 32MB) | Ch(f16 16MB) | Xh(f16 8MB)
  unsigned*  Cc = (unsigned*)d_ws;
  _Float16*  Ch = (_Float16*)((char*)d_ws + (size_t)33554432);
  _Float16*  Xh = (_Float16*)((char*)d_ws + (size_t)33554432 + (size_t)16777216);

  const int nC = BG * NN * NN;     // 8388608
  const int nX = BG * NN * DD;     // 4194304
  const int nE = in_sizes[1] / 2;  // 524288 directed edges

  zero_u32_k     <<<(nC + 255) / 256, 256, 0, stream>>>(Cc, nC);
  scatter_edges_k<<<(nE + 255) / 256, 256, 0, stream>>>(EI, Cc, nE);
  cvt_count_f16_k<<<(nC + 255) / 256, 256, 0, stream>>>(Cc, Ch, nC);
  cvt_f32_f16_k  <<<(nX + 255) / 256, 256, 0, stream>>>(X, Xh, nX);
  tfgw_main_k    <<<BG * KT, 256, 0, stream>>>(Xh, Ch, X, Adj, Feat, q0, alpha0, out);
}
